// FamModelMoELSTM_13357348291022
// MI455X (gfx1250) — matrix operand order repaired and checked by reading, not compiled, not run on hardware
//
#include <hip/hip_runtime.h>
#include <hip/hip_bf16.h>

typedef __attribute__((ext_vector_type(16))) _Float16 v16h;
typedef __attribute__((ext_vector_type(8)))  float    v8f;
typedef __attribute__((ext_vector_type(2)))  _Float16 h2;
typedef __attribute__((ext_vector_type(8)))  _Float16 h8;

#define TOK_N 8192
#define EMB_D 512
#define HID_H 512
#define GATE4 2048
#define CLANS 16
#define FAM   64

// ---------------- WMMA tile loaders (CDNA5 16-bit layouts, ISA 7.12.2) ----
// A 16x32 f16: lane&15 = M row; VGPR p<4 -> K = kh + 2p{,+1}; p>=4 -> +16
__device__ __forceinline__ v16h load_tileA(const _Float16* __restrict__ rowbase,
                                           int k0, int lane) {
  v16h a;
  const int kh = (lane & 16) ? 8 : 0;
#pragma unroll
  for (int p = 0; p < 8; ++p) {
    int k = k0 + ((p & 4) ? 16 : 0) + kh + (p & 3) * 2;
    h2 v = *(const h2*)(rowbase + k);
    a[2 * p]     = v.x;
    a[2 * p + 1] = v.y;
  }
  return a;
}
// B 32x16 f16 (loaded from k-contiguous row of B^T): lane&15 = N col;
// K = (lane>=16 ? 8 : 0) + 2p{,+1} for p<4; p>=4 -> +16 (same as A)
__device__ __forceinline__ v16h load_tileB(const _Float16* __restrict__ rowbase,
                                           int k0, int lane) {
  v16h b;
  const int kh = (lane & 16) ? 8 : 0;
#pragma unroll
  for (int p = 0; p < 8; ++p) {
    int k = k0 + ((p & 4) ? 16 : 0) + kh + (p & 3) * 2;
    h2 v = *(const h2*)(rowbase + k);
    b[2 * p]     = v.x;
    b[2 * p + 1] = v.y;
  }
  return b;
}

// ---------------- utility kernels ----------------------------------------
__global__ void k_zero(float* __restrict__ out, size_t n,
                       int* __restrict__ counts, int* __restrict__ slots) {
  size_t i = (size_t)blockIdx.x * blockDim.x + threadIdx.x;
  if (i < n) out[i] = 0.f;
  if (i < CLANS) { counts[i] = 0; slots[i] = 0; }
}

__global__ void k_f32_to_f16(const float* __restrict__ src,
                             _Float16* __restrict__ dst, int n) {
  int i = blockIdx.x * blockDim.x + threadIdx.x;
  if (i < n) dst[i] = (_Float16)src[i];
}

// pack Whh (4H x H, gate-major rows i,f,g,o) -> [j][k][gate] f16 for b128 streaming
__global__ void k_pack_whh(const float* __restrict__ Whh,
                           _Float16* __restrict__ Wp) {
  int i = blockIdx.x * blockDim.x + threadIdx.x;  // over 512*512*4
  if (i >= HID_H * HID_H * 4) return;
  int g = i & 3;
  int k = (i >> 2) & (HID_H - 1);
  int j = i >> 11;
  Wp[i] = (_Float16)Whh[(g * HID_H + j) * HID_H + k];
}

// W1 (C,1024,128) -> W1T f16 (C,128,1024)  [n-major rows, k contiguous]
__global__ void k_tr_w1(const float* __restrict__ W1,
                        _Float16* __restrict__ W1T) {
  int i = blockIdx.x * blockDim.x + threadIdx.x;  // dst linear
  if (i >= CLANS * 128 * 1024) return;
  int k = i & 1023;
  int n = (i >> 10) & 127;
  int c = i >> 17;
  W1T[i] = (_Float16)W1[(c * 1024 + k) * 128 + n];
}
// W2 (C,128,64) -> W2T f16 (C,64,128)
__global__ void k_tr_w2(const float* __restrict__ W2,
                        _Float16* __restrict__ W2T) {
  int i = blockIdx.x * blockDim.x + threadIdx.x;
  if (i >= CLANS * 64 * 128) return;
  int k = i & 127;
  int f = (i >> 7) & 63;
  int c = i >> 13;
  W2T[i] = (_Float16)W2[(c * 128 + k) * 64 + f];
}

// ---------------- input projection GEMM (WMMA) ---------------------------
// xz[N,4H] = x[N,D] @ Wih^T + b   for both directions (blockIdx.z)
__global__ __launch_bounds__(128) void k_xproj(
    const _Float16* __restrict__ x16,
    const _Float16* __restrict__ Wf, const _Float16* __restrict__ Wb,
    const float* __restrict__ bf, const float* __restrict__ bb,
    float* __restrict__ xzf, float* __restrict__ xzb) {
  const int lane = threadIdx.x & 31;
  const int wave = threadIdx.x >> 5;
  const _Float16* W = blockIdx.z ? Wb : Wf;
  const float* bias = blockIdx.z ? bb : bf;
  float* out = blockIdx.z ? xzb : xzf;

  const int m0 = blockIdx.x * 16;
  const int n0 = (blockIdx.y * 4 + wave) * 16;
  const _Float16* arow = x16 + (size_t)(m0 + (lane & 15)) * EMB_D;
  const _Float16* brow = W + (size_t)(n0 + (lane & 15)) * EMB_D;  // Wih[n][k]

  v8f c = {};
  for (int k0 = 0; k0 < EMB_D; k0 += 32) {
    v16h a = load_tileA(arow, k0, lane);
    v16h b = load_tileB(brow, k0, lane);
    c = __builtin_amdgcn_wmma_f32_16x16x32_f16(false, a, false, b, (short)0,
                                               c, false, false);
  }
  const int n = n0 + (lane & 15);
  const float bv = bias[n];
  const int mb = (lane & 16) ? 8 : 0;
#pragma unroll
  for (int r = 0; r < 8; ++r)
    out[(size_t)(m0 + mb + r) * GATE4 + n] = c[r] + bv;
}

// ---------------- sequential BiLSTM scan ---------------------------------
// one persistent workgroup per direction; Whh packed f16 lives in L2
__global__ __launch_bounds__(512) void k_lstm(
    const _Float16* __restrict__ WpF, const _Float16* __restrict__ WpB,
    const float* __restrict__ xzf, const float* __restrict__ xzb,
    float* __restrict__ hcat) {
  const int dir = blockIdx.x;
  const int j = threadIdx.x;  // hidden index 0..511
  const _Float16* Wp = (dir ? WpB : WpF) + (size_t)j * (HID_H * 4);
  const float* xz = dir ? xzb : xzf;

  __shared__ __align__(16) _Float16 hsh[HID_H];
  hsh[j] = (_Float16)0.f;
  __syncthreads();

  float c = 0.f;
  for (int s = 0; s < TOK_N; ++s) {
    const int t = dir ? (TOK_N - 1 - s) : s;
    const float* z = xz + (size_t)t * GATE4;
    // prefetch next step's gate preactivations into cache
    const int tn = dir ? (t - 1) : (t + 1);
    if (tn >= 0 && tn < TOK_N) __builtin_prefetch(xz + (size_t)tn * GATE4, 0, 0);

    float ai = z[j], af = z[HID_H + j], ag = z[2 * HID_H + j],
          ao = z[3 * HID_H + j];
#pragma unroll 4
    for (int kk = 0; kk < HID_H; kk += 2) {
      h2 hv = *(const h2*)&hsh[kk];
      h8 w = *(const h8*)(Wp + (size_t)kk * 4);  // 16B: gates i,f,g,o for k,k+1
      float h0 = (float)hv.x, h1 = (float)hv.y;
      ai = fmaf((float)w[0], h0, ai);
      af = fmaf((float)w[1], h0, af);
      ag = fmaf((float)w[2], h0, ag);
      ao = fmaf((float)w[3], h0, ao);
      ai = fmaf((float)w[4], h1, ai);
      af = fmaf((float)w[5], h1, af);
      ag = fmaf((float)w[6], h1, ag);
      ao = fmaf((float)w[7], h1, ao);
    }
    float ig = 1.f / (1.f + __expf(-ai));
    float fg = 1.f / (1.f + __expf(-af));
    float gg = tanhf(ag);
    float og = 1.f / (1.f + __expf(-ao));
    c = fg * c + ig * gg;
    float h = og * tanhf(c);
    __syncthreads();               // all reads of old h done
    hsh[j] = (_Float16)h;
    hcat[(size_t)t * (2 * HID_H) + dir * HID_H + j] = h;
    __syncthreads();               // new h visible
  }
}

// ---------------- routing ------------------------------------------------
__global__ void k_route(const float* __restrict__ xc, int* __restrict__ clan,
                        int* __restrict__ counts, int n) {
  int i = blockIdx.x * blockDim.x + threadIdx.x;
  if (i >= n) return;
  const float* r = xc + (size_t)i * CLANS;
  int best = 0;
  float bv = r[0];
#pragma unroll
  for (int c = 1; c < CLANS; ++c) {
    float v = r[c];
    if (v > bv) { bv = v; best = c; }  // strict > keeps first (argmax tie rule)
  }
  clan[i] = best;
  atomicAdd(&counts[best], 1);
}

__global__ void k_scan(const int* __restrict__ counts, int* __restrict__ offs) {
  if (threadIdx.x == 0) {
    int acc = 0;
    for (int c = 0; c < CLANS; ++c) { offs[c] = acc; acc += counts[c]; }
  }
}

__global__ void k_scatter(const int* __restrict__ clan,
                          const int* __restrict__ offs, int* __restrict__ slots,
                          int* __restrict__ idxl, int n) {
  int i = blockIdx.x * blockDim.x + threadIdx.x;
  if (i >= n) return;
  int c = clan[i];
  int p = atomicAdd(&slots[c], 1);
  idxl[offs[c] + p] = i;
}

// ---------------- MoE expert MLP (WMMA, gathered 16-token tiles) ----------
__global__ __launch_bounds__(128) void k_moe(
    const _Float16* __restrict__ h16,   // N x 1024
    const _Float16* __restrict__ W1T,   // C x 128 x 1024
    const float* __restrict__ b1,       // C x 128
    const float* __restrict__ gmm, const float* __restrict__ bta,
    const _Float16* __restrict__ W2T,   // C x 64 x 128
    const float* __restrict__ b2,       // C x 64
    const int* __restrict__ counts, const int* __restrict__ offs,
    const int* __restrict__ idxl, float* __restrict__ out) {
  __shared__ float s_h1[4][16][132];                 // padded: conflict-free
  __shared__ __align__(16) _Float16 s_a2[4][16][136];
  __shared__ int s_tok[4][16];

  const int lane = threadIdx.x & 31;
  const int wave = threadIdx.x >> 5;
  const int e = blockIdx.x;
  const int tile = blockIdx.y * 4 + wave;
  const int cnt = counts[e];
  const bool tile_active = (tile * 16 < cnt);        // wave-uniform

  const int m = lane & 15;
  const int row = tile * 16 + m;
  const bool row_valid = tile_active && (row < cnt);
  const int tok = row_valid ? idxl[offs[e] + row] : 0;
  if (lane < 16) s_tok[wave][lane] = row_valid ? tok : -1;
  const int mb = (lane & 16) ? 8 : 0;

  if (tile_active) {
    // ---- GEMM1: [16 x 1024] @ [1024 x 128] ----
    v8f zero = {};
    v8f acc[8];
#pragma unroll
    for (int nt = 0; nt < 8; ++nt) acc[nt] = zero;
    const _Float16* arow = h16 + (size_t)tok * 1024;
    const _Float16* bbase = W1T + ((size_t)e * 128 + m) * 1024;
    for (int k0 = 0; k0 < 1024; k0 += 32) {
      v16h a = load_tileA(arow, k0, lane);
#pragma unroll
      for (int nt = 0; nt < 8; ++nt) {
        v16h b = load_tileB(bbase + (size_t)nt * 16 * 1024, k0, lane);
        acc[nt] = __builtin_amdgcn_wmma_f32_16x16x32_f16(
            false, a, false, b, (short)0, acc[nt], false, false);
      }
    }
#pragma unroll
    for (int nt = 0; nt < 8; ++nt) {
      int n = nt * 16 + m;
      float bv = b1[e * 128 + n];
#pragma unroll
      for (int r = 0; r < 8; ++r) s_h1[wave][mb + r][n] = acc[nt][r] + bv;
    }
  }
  __syncthreads();

  if (tile_active && lane < 16) {
    // ---- LayerNorm + ReLU over 128 per token row ----
    float sum = 0.f, sq = 0.f;
#pragma unroll 8
    for (int n = 0; n < 128; ++n) {
      float v = s_h1[wave][lane][n];
      sum += v; sq += v * v;
    }
    float mu = sum * (1.f / 128.f);
    float var = sq * (1.f / 128.f) - mu * mu;
    float rs = rsqrtf(var + 1e-5f);
#pragma unroll 8
    for (int n = 0; n < 128; ++n) {
      float v = (s_h1[wave][lane][n] - mu) * rs * gmm[e * 128 + n] +
                bta[e * 128 + n];
      s_a2[wave][lane][n] = (_Float16)(v > 0.f ? v : 0.f);
    }
  }
  __syncthreads();

  if (tile_active) {
    // ---- GEMM2: [16 x 128] @ [128 x 64] ----
    v8f zero = {};
    v8f acc2[4];
#pragma unroll
    for (int nt = 0; nt < 4; ++nt) acc2[nt] = zero;
    const _Float16* a2row = &s_a2[wave][m][0];
    const _Float16* b2base = W2T + ((size_t)e * 64 + m) * 128;
    for (int k0 = 0; k0 < 128; k0 += 32) {
      v16h a = load_tileA(a2row, k0, lane);
#pragma unroll
      for (int nt = 0; nt < 4; ++nt) {
        v16h b = load_tileB(b2base + (size_t)nt * 16 * 128, k0, lane);
        acc2[nt] = __builtin_amdgcn_wmma_f32_16x16x32_f16(
            false, a, false, b, (short)0, acc2[nt], false, false);
      }
    }
#pragma unroll
    for (int nt = 0; nt < 4; ++nt) {
      int f = nt * 16 + m;
      float bv = b2[e * 64 + f];
#pragma unroll
      for (int r = 0; r < 8; ++r) {
        int tk = s_tok[wave][mb + r];
        if (tk >= 0)
          out[(size_t)tk * (CLANS * FAM) + e * FAM + f] = acc2[nt][r] + bv;
      }
    }
  }
}

// ---------------- launch -------------------------------------------------
extern "C" void kernel_launch(void* const* d_in, const int* in_sizes, int n_in,
                              void* d_out, int out_size, void* d_ws,
                              size_t ws_size, hipStream_t stream) {
  const float* x    = (const float*)d_in[0];
  const float* xc   = (const float*)d_in[1];
  const float* Wihf = (const float*)d_in[2];
  const float* Whhf = (const float*)d_in[3];
  const float* bf   = (const float*)d_in[4];
  const float* Wihb = (const float*)d_in[5];
  const float* Whhb = (const float*)d_in[6];
  const float* bb   = (const float*)d_in[7];
  const float* W1   = (const float*)d_in[8];
  const float* b1   = (const float*)d_in[9];
  const float* gmm  = (const float*)d_in[10];
  const float* bta  = (const float*)d_in[11];
  const float* W2   = (const float*)d_in[12];
  const float* b2   = (const float*)d_in[13];
  float* out = (float*)d_out;

  char* ws = (char*)d_ws;
  size_t off = 0;
  auto alloc = [&](size_t bytes) -> void* {
    void* p = ws + off;
    off = (off + bytes + 255) & ~(size_t)255;
    return p;
  };
  _Float16* x16  = (_Float16*)alloc((size_t)TOK_N * EMB_D * 2);
  _Float16* Wf16 = (_Float16*)alloc((size_t)GATE4 * EMB_D * 2);
  _Float16* Wb16 = (_Float16*)alloc((size_t)GATE4 * EMB_D * 2);
  _Float16* WpF  = (_Float16*)alloc((size_t)HID_H * HID_H * 4 * 2);
  _Float16* WpB  = (_Float16*)alloc((size_t)HID_H * HID_H * 4 * 2);
  _Float16* W1T  = (_Float16*)alloc((size_t)CLANS * 128 * 1024 * 2);
  _Float16* W2T  = (_Float16*)alloc((size_t)CLANS * 64 * 128 * 2);
  float*    xzf  = (float*)alloc((size_t)TOK_N * GATE4 * 4);
  float*    xzb  = (float*)alloc((size_t)TOK_N * GATE4 * 4);
  float*    hcat = (float*)alloc((size_t)TOK_N * 1024 * 4);
  _Float16* h16  = (_Float16*)alloc((size_t)TOK_N * 1024 * 2);
  int* clan   = (int*)alloc(TOK_N * 4);
  int* counts = (int*)alloc(64);
  int* offs   = (int*)alloc(64);
  int* slots  = (int*)alloc(64);
  int* idxl   = (int*)alloc(TOK_N * 4);

  // zero output + counters (re-done each call; deterministic)
  k_zero<<<(out_size + 255) / 256, 256, 0, stream>>>((float*)out,
                                                     (size_t)out_size, counts,
                                                     slots);
  // precision/layout conversions
  k_f32_to_f16<<<(TOK_N * EMB_D + 255) / 256, 256, 0, stream>>>(x, x16,
                                                                TOK_N * EMB_D);
  k_f32_to_f16<<<(GATE4 * EMB_D + 255) / 256, 256, 0, stream>>>(
      Wihf, Wf16, GATE4 * EMB_D);
  k_f32_to_f16<<<(GATE4 * EMB_D + 255) / 256, 256, 0, stream>>>(
      Wihb, Wb16, GATE4 * EMB_D);
  k_pack_whh<<<(HID_H * HID_H * 4 + 255) / 256, 256, 0, stream>>>(Whhf, WpF);
  k_pack_whh<<<(HID_H * HID_H * 4 + 255) / 256, 256, 0, stream>>>(Whhb, WpB);
  k_tr_w1<<<(CLANS * 128 * 1024 + 255) / 256, 256, 0, stream>>>(W1, W1T);
  k_tr_w2<<<(CLANS * 64 * 128 + 255) / 256, 256, 0, stream>>>(W2, W2T);

  // input-projection GEMMs (WMMA), both directions
  k_xproj<<<dim3(TOK_N / 16, GATE4 / 64, 2), 128, 0, stream>>>(
      x16, Wf16, Wb16, bf, bb, xzf, xzb);

  // sequential scan, 2 persistent workgroups (fwd / bwd)
  k_lstm<<<2, 512, 0, stream>>>(WpF, WpB, xzf, xzb, hcat);

  k_f32_to_f16<<<(TOK_N * 1024 + 255) / 256, 256, 0, stream>>>(hcat, h16,
                                                               TOK_N * 1024);
  // routing: argmax -> counts -> offsets -> bucketed index list
  k_route<<<(TOK_N + 255) / 256, 256, 0, stream>>>(xc, clan, counts, TOK_N);
  k_scan<<<1, 32, 0, stream>>>(counts, offs);
  k_scatter<<<(TOK_N + 255) / 256, 256, 0, stream>>>(clan, offs, slots, idxl,
                                                     TOK_N);
  // expert MLPs on gathered 16-token tiles (WMMA)
  k_moe<<<dim3(CLANS, 128), 128, 0, stream>>>(h16, W1T, b1, gmm, bta, W2T, b2,
                                              counts, offs, idxl, out);
  (void)in_sizes; (void)n_in; (void)ws_size;
}
